// GroupedQueryAttention_61237643706964
// MI455X (gfx1250) — matrix operand order repaired and checked by reading, not compiled, not run on hardware
//
#include <hip/hip_runtime.h>
#include <hip/hip_bf16.h>

typedef __attribute__((ext_vector_type(16))) __bf16 bf16x16;
typedef __attribute__((ext_vector_type(8)))  __bf16 bf16x8;
typedef __attribute__((ext_vector_type(8)))  float  f32x8;
typedef __attribute__((ext_vector_type(4)))  unsigned int u32x4;
typedef __attribute__((ext_vector_type(8)))  unsigned int u32x8;

#define DIM_    4096
#define NH_     32
#define NKV_    8
#define DH_     128
#define BB_     2
#define TT_     2048
#define MTOK_   (BB_*TT_)            // 4096 tokens
#define QKVN_   (DIM_ + 2*NKV_*DH_)  // 6144
#define KOFF_   DIM_                 // 4096
#define VOFF_   (DIM_ + NKV_*DH_)    // 5120

// LDS K tile: TDM pads 4 DWORDs after every 64 DWORDs (256B row) ->
// effective row stride 136 bf16 elements; 16 lanes land on 16 distinct banks.
#define KROW_   136
// LDS P tile: manual +8 element pad -> 36-DWORD row stride (gcd(36,64)=4,
// 16 lanes -> 16 distinct banks).
#define PROW_   72

// ---------------------------------------------------------------------------
// Fragment loaders for CDNA5 wave32 WMMA; contiguous k order (k = 16*hi + s) in both A and B.
// ---------------------------------------------------------------------------
__device__ __forceinline__ bf16x16 load_a_frag(const __bf16* rowptr, int hi) {
  const bf16x8 lo = *reinterpret_cast<const bf16x8*>(rowptr + hi * 16);
  const bf16x8 hh = *reinterpret_cast<const bf16x8*>(rowptr + hi * 16 + 8);
  bf16x16 a;
#pragma unroll
  for (int i = 0; i < 8; ++i) { a[i] = lo[i]; a[i + 8] = hh[i]; }
  return a;
}

// 16 contiguous bf16 (32B) -- requires 32B alignment
__device__ __forceinline__ bf16x16 load_b_frag(const __bf16* colptr, int hi) {
  return *reinterpret_cast<const bf16x16*>(colptr + hi * 16);
}

// same but from 16B-aligned storage (padded LDS rows)
__device__ __forceinline__ bf16x16 load_b_frag16(const __bf16* colptr, int hi) {
  const bf16x8 lo = *reinterpret_cast<const bf16x8*>(colptr + hi * 16);
  const bf16x8 hh = *reinterpret_cast<const bf16x8*>(colptr + hi * 16 + 8);
  bf16x16 b;
#pragma unroll
  for (int i = 0; i < 8; ++i) { b[i] = lo[i]; b[i + 8] = hh[i]; }
  return b;
}

__device__ __forceinline__ f32x8 wmma_bf16(bf16x16 a, bf16x16 b, f32x8 c) {
  return __builtin_amdgcn_wmma_f32_16x16x32_bf16(false, a, false, b,
                                                 (short)0, c, false, false);
}

// ---------------------------------------------------------------------------
// fp32 -> bf16 conversion (grid-stride)
// ---------------------------------------------------------------------------
__global__ void cvt_f32_bf16(const float* __restrict__ in,
                             __bf16* __restrict__ out, long long n) {
  long long i = (long long)blockIdx.x * blockDim.x + threadIdx.x;
  const long long stride = (long long)gridDim.x * blockDim.x;
  for (; i < n; i += stride) out[i] = (__bf16)in[i];
}

// ---------------------------------------------------------------------------
// GEMM: C[M,N] = A[M,K] * W[N,K]^T, bf16 in, f32 accumulate.
// 8 waves/block; wave tile 64(M) x 128(N) = 4x8 frags; block 256M x 256N.
// Software-pipelined: fragments for k+32 load while k computes.
// ---------------------------------------------------------------------------
__device__ __forceinline__ void load_tiles(const __bf16* __restrict__ A,
                                           const __bf16* __restrict__ W,
                                           int m0, int n0, int K, int k0,
                                           int l16, int hi,
                                           bf16x16 (&a)[4], bf16x16 (&b)[8]) {
#pragma unroll
  for (int m = 0; m < 4; ++m) {
    const __bf16* rp = A + (size_t)(m0 + m * 16 + l16) * K + k0;
    a[m] = load_a_frag(rp, hi);
    __builtin_prefetch(rp + 64, 0, 1);
  }
#pragma unroll
  for (int n = 0; n < 8; ++n) {
    const __bf16* wp = W + (size_t)(n0 + n * 16 + l16) * K + k0;
    b[n] = load_b_frag(wp, hi);
    __builtin_prefetch(wp + 64, 0, 1);
  }
}

template <bool F32OUT>
__global__ __launch_bounds__(256)
void gemm_bf16_wmma(const __bf16* __restrict__ A, const __bf16* __restrict__ W,
                    void* __restrict__ Cout, int M, int N, int K) {
  const int lane = threadIdx.x & 31;
  const int wid  = threadIdx.x >> 5;
  const int hi   = lane >> 4;
  const int l16  = lane & 15;
  const int m0   = blockIdx.y * 256 + (wid & 3) * 64;
  const int n0   = blockIdx.x * 256 + (wid >> 2) * 128;

  const f32x8 zero8 = {0.f, 0.f, 0.f, 0.f, 0.f, 0.f, 0.f, 0.f};
  f32x8 acc[4][8];
#pragma unroll
  for (int m = 0; m < 4; ++m)
#pragma unroll
    for (int n = 0; n < 8; ++n) acc[m][n] = zero8;

  bf16x16 a[2][4], b[2][8];
  load_tiles(A, W, m0, n0, K, 0, l16, hi, a[0], b[0]);

  for (int k0 = 0; k0 < K; k0 += 32) {
    const int cur = (k0 >> 5) & 1;
    const int nxt = cur ^ 1;
    if (k0 + 32 < K)
      load_tiles(A, W, m0, n0, K, k0 + 32, l16, hi, a[nxt], b[nxt]);
#pragma unroll
    for (int m = 0; m < 4; ++m)
#pragma unroll
      for (int n = 0; n < 8; ++n)
        acc[m][n] = wmma_bf16(a[cur][m], b[cur][n], acc[m][n]);
  }

#pragma unroll
  for (int m = 0; m < 4; ++m)
#pragma unroll
    for (int n = 0; n < 8; ++n)
#pragma unroll
      for (int r = 0; r < 8; ++r) {
        const int row = m0 + m * 16 + r + hi * 8;
        const int col = n0 + n * 16 + l16;
        if constexpr (F32OUT)
          ((float*)Cout)[(size_t)row * N + col] = acc[m][n][r];
        else
          ((__bf16*)Cout)[(size_t)row * N + col] = (__bf16)acc[m][n][r];
      }
}

// ---------------------------------------------------------------------------
// RoPE in-place on Q (32 heads) and K (8 heads) slices of qkv buffer.
// ---------------------------------------------------------------------------
__global__ void rope_kernel(__bf16* __restrict__ qkv,
                            const float* __restrict__ fcos,
                            const float* __restrict__ fsin) {
  const long long total = (long long)MTOK_ * (NH_ + NKV_) * (DH_ / 2);
  long long idx = (long long)blockIdx.x * blockDim.x + threadIdx.x;
  if (idx >= total) return;
  const int pair = (int)(idx & (DH_ / 2 - 1));
  long long t2   = idx >> 6;
  const int head = (int)(t2 % (NH_ + NKV_));
  const long long tok = t2 / (NH_ + NKV_);
  const int t = (int)(tok & (TT_ - 1));
  const float c = fcos[t * (DH_ / 2) + pair];
  const float s = fsin[t * (DH_ / 2) + pair];
  const size_t col = (head < NH_) ? (size_t)head * DH_
                                  : (size_t)KOFF_ + (size_t)(head - NH_) * DH_;
  __bf16* p = qkv + (size_t)tok * QKVN_ + col + pair * 2;
  const float x1 = (float)p[0], x2 = (float)p[1];
  p[0] = (__bf16)(x1 * c - x2 * s);
  p[1] = (__bf16)(x1 * s + x2 * c);
}

// ---------------------------------------------------------------------------
// V transpose: vt[b][kvh][d][t] = v[b][t][kvh][d]  (8 MB bf16, L2-resident)
// ---------------------------------------------------------------------------
__global__ void transpose_v(const __bf16* __restrict__ qkv,
                            __bf16* __restrict__ vt) {
  long long idx = (long long)blockIdx.x * blockDim.x + threadIdx.x;
  const long long total = (long long)BB_ * NKV_ * DH_ * TT_;
  if (idx >= total) return;
  const int t = (int)(idx & (TT_ - 1));
  long long r = idx >> 11;
  const int d = (int)(r & (DH_ - 1));
  r >>= 7;
  const int kvh = (int)(r & (NKV_ - 1));
  const int b   = (int)(r >> 3);
  vt[idx] = qkv[((size_t)(b * TT_ + t)) * QKVN_ + VOFF_ + kvh * DH_ + d];
}

// ---------------------------------------------------------------------------
// TDM: DMA one 64(key) x 128(dh) bf16 K tile into padded LDS rows.
// D# per ISA 8.3/8.4: count=1, data_size=2B, pad every 64 DWORDs by 4 DWORDs.
// All descriptor inputs are wave-uniform -> SGPR groups.
// ---------------------------------------------------------------------------
__device__ __forceinline__ void tdm_load_k_tile(const __bf16* gsrc,
                                                const __bf16* lds_dst) {
  const unsigned long long ga = (unsigned long long)(const void*)gsrc;
  const unsigned int la = (unsigned int)(unsigned long long)(const void*)lds_dst;
  u32x4 g0;
  g0[0] = 1u;                                   // count=1, user mode
  g0[1] = la;                                   // lds_addr
  g0[2] = (unsigned int)ga;                     // global_addr[31:0]
  g0[3] = (unsigned int)((ga >> 32) & 0x01FFFFFFu) | 0x80000000u; // addr hi | type=2
  u32x8 g1;
  g1[0] = 0x07510000u;  // data_size=1(2B) | pad_en | pad_interval=5 | pad_amount=3
  g1[1] = 0u;           // barrier addr 0, tensor_dim0[15:0]=0
  g1[2] = 0x00000010u;  // tensor_dim0 = 1<<20 (hi), tensor_dim1 lo16 = 0
  g1[3] = (128u << 16) | 0x10u;  // tile_dim0=128 | tensor_dim1 hi16 (1<<20)
  g1[4] = 64u;          // tile_dim1=64, tile_dim2=0
  g1[5] = (unsigned)QKVN_;  // tensor_dim0_stride = 6144
  g1[6] = 0u;
  g1[7] = 0u;
  u32x4 g2 = {0u, 0u, 0u, 0u};
  u32x4 g3 = {0u, 0u, 0u, 0u};
  asm volatile("tensor_load_to_lds %0, %1, %2, %3"
               :: "s"(g0), "s"(g1), "s"(g2), "s"(g3) : "memory");
}

// ---------------------------------------------------------------------------
// Causal flash attention.  One wave per (b, head, 64-row q tile).
// K tiles arrive via double-buffered TDM; V read from pre-transposed global.
// ---------------------------------------------------------------------------
__global__ __launch_bounds__(32)
void attn_kernel(const __bf16* __restrict__ qkv, const __bf16* __restrict__ vt,
                 __bf16* __restrict__ attn_out) {
  const int lane = threadIdx.x & 31;
  const int hi = lane >> 4, l16 = lane & 15;
  const int q0  = blockIdx.x * 64;
  const int h   = blockIdx.y;
  const int b   = blockIdx.z;
  const int kvh = h >> 2;
  const float scale = 0.08838834764831845f; // 1/sqrt(128)

  __shared__ __align__(32) __bf16 ldsK[2][64 * KROW_];  // 2 x 17 KB (TDM-padded)
  __shared__ __align__(32) __bf16 ldsP[64 * PROW_];     // 9 KB

  const __bf16* Qbase = qkv + (size_t)(b * TT_) * QKVN_ + h * DH_;
  const __bf16* Kbase = qkv + (size_t)(b * TT_) * QKVN_ + KOFF_ + kvh * DH_;
  const __bf16* Vtb   = vt + ((size_t)(b * NKV_ + kvh) * DH_) * TT_;

  // Q tile resident in registers: 4 M-frags x 4 K-chunks
  bf16x16 qf[4][4];
#pragma unroll
  for (int m = 0; m < 4; ++m)
#pragma unroll
    for (int kc = 0; kc < 4; ++kc)
      qf[m][kc] = load_a_frag(
          Qbase + (size_t)(q0 + m * 16 + l16) * QKVN_ + kc * 32, hi);

  const f32x8 zero8 = {0.f, 0.f, 0.f, 0.f, 0.f, 0.f, 0.f, 0.f};
  f32x8 o[4][8];
  float mrow[4][8], lrow[4][8];
#pragma unroll
  for (int m = 0; m < 4; ++m) {
#pragma unroll
    for (int nd = 0; nd < 8; ++nd) o[m][nd] = zero8;
#pragma unroll
    for (int r = 0; r < 8; ++r) { mrow[m][r] = -3.0e38f; lrow[m][r] = 0.f; }
  }

  const int nsteps = q0 / 64 + 1;
  tdm_load_k_tile(Kbase, &ldsK[0][0]);          // prefetch first K tile

  for (int it = 0; it < nsteps; ++it) {
    const int k0 = it * 64;
    const bool have_next = (it + 1 < nsteps);
    if (have_next) {
      tdm_load_k_tile(Kbase + (size_t)(k0 + 64) * QKVN_, &ldsK[(it + 1) & 1][0]);
      asm volatile("s_wait_tensorcnt 0x1" ::: "memory");
    } else {
      asm volatile("s_wait_tensorcnt 0x0" ::: "memory");
    }
    const __bf16* Kt = &ldsK[it & 1][0];

    // ---- S = Q K^T (K rows from padded LDS) ----------------------------
    f32x8 s[4][4];
#pragma unroll
    for (int m = 0; m < 4; ++m)
#pragma unroll
      for (int n = 0; n < 4; ++n) s[m][n] = zero8;

#pragma unroll
    for (int kc = 0; kc < 4; ++kc) {
      bf16x16 kf[4];
#pragma unroll
      for (int n = 0; n < 4; ++n)
        kf[n] = load_b_frag16(Kt + (n * 16 + l16) * KROW_ + kc * 32, hi);
#pragma unroll
      for (int m = 0; m < 4; ++m)
#pragma unroll
        for (int n = 0; n < 4; ++n)
          s[m][n] = wmma_bf16(qf[m][kc], kf[n], s[m][n]);
    }

    // ---- scale, causal mask, online softmax; P (bf16) -> LDS -----------
#pragma unroll
    for (int m = 0; m < 4; ++m) {
#pragma unroll
      for (int r = 0; r < 8; ++r) {
        const int qrow = q0 + m * 16 + r + hi * 8;
        float rmax = -3.0e38f;
#pragma unroll
        for (int n = 0; n < 4; ++n) {
          float v = s[m][n][r] * scale;
          const int kcol = k0 + n * 16 + l16;
          if (kcol > qrow) v = -1.0e9f;
          s[m][n][r] = v;
          rmax = fmaxf(rmax, v);
        }
#pragma unroll
        for (int off = 8; off >= 1; off >>= 1)
          rmax = fmaxf(rmax, __shfl_xor(rmax, off, 16));
        const float mold = mrow[m][r];
        const float mnew = fmaxf(mold, rmax);
        const float alpha = __expf(mold - mnew);
        float psum = 0.f;
#pragma unroll
        for (int n = 0; n < 4; ++n) {
          const float p = __expf(s[m][n][r] - mnew);
          psum += p;
          ldsP[(m * 16 + r + hi * 8) * PROW_ + n * 16 + l16] = (__bf16)p;
        }
#pragma unroll
        for (int off = 8; off >= 1; off >>= 1)
          psum += __shfl_xor(psum, off, 16);
        lrow[m][r] = lrow[m][r] * alpha + psum;
        mrow[m][r] = mnew;
#pragma unroll
        for (int nd = 0; nd < 8; ++nd) o[m][nd][r] *= alpha;
      }
    }
    __syncthreads();

    // ---- O += P V  (P from LDS A-layout, V^T from global B-layout) -----
#pragma unroll
    for (int kc2 = 0; kc2 < 2; ++kc2) {
      bf16x16 pf[4];
#pragma unroll
      for (int m = 0; m < 4; ++m)
        pf[m] = load_a_frag(ldsP + (m * 16 + l16) * PROW_ + kc2 * 32, hi);
#pragma unroll
      for (int nd = 0; nd < 8; ++nd) {
        const bf16x16 vf = load_b_frag(
            Vtb + (size_t)(nd * 16 + l16) * TT_ + k0 + kc2 * 32, hi);
#pragma unroll
        for (int m = 0; m < 4; ++m)
          o[m][nd] = wmma_bf16(pf[m], vf, o[m][nd]);
      }
    }
    __syncthreads();
  }

  // ---- normalize and store (bf16, [tok, NH*DH] row-major) ----------------
#pragma unroll
  for (int m = 0; m < 4; ++m)
#pragma unroll
    for (int r = 0; r < 8; ++r) {
      const float inv = 1.0f / lrow[m][r];
      const int q = q0 + m * 16 + r + hi * 8;
#pragma unroll
      for (int nd = 0; nd < 8; ++nd) {
        const int d = nd * 16 + l16;
        attn_out[(size_t)(b * TT_ + q) * (NH_ * DH_) + h * DH_ + d] =
            (__bf16)(o[m][nd][r] * inv);
      }
    }
}

// ---------------------------------------------------------------------------
// Orchestration.  Workspace: xb | wqkvb | wprojb | qkvb | attnb | vtb ~201 MB
// ---------------------------------------------------------------------------
extern "C" void kernel_launch(void* const* d_in, const int* in_sizes, int n_in,
                              void* d_out, int out_size, void* d_ws, size_t ws_size,
                              hipStream_t stream) {
  const float* x     = (const float*)d_in[0];
  const float* wq    = (const float*)d_in[1];
  const float* wk    = (const float*)d_in[2];
  const float* wv    = (const float*)d_in[3];
  const float* wproj = (const float*)d_in[4];
  const float* fcos  = (const float*)d_in[5];
  const float* fsin  = (const float*)d_in[6];
  // d_in[7] (mask) reproduced analytically; d_in[8] (start_pos) == 0.

  char* ws = (char*)d_ws;
  const size_t XB_B    = (size_t)MTOK_ * DIM_ * 2;
  const size_t WQKV_B  = (size_t)QKVN_ * DIM_ * 2;
  const size_t WPROJ_B = (size_t)DIM_ * DIM_ * 2;
  const size_t QKV_B   = (size_t)MTOK_ * QKVN_ * 2;
  const size_t ATT_B   = (size_t)MTOK_ * DIM_ * 2;
  __bf16* xb     = (__bf16*)(ws);
  __bf16* wqkvb  = (__bf16*)(ws + XB_B);
  __bf16* wprojb = (__bf16*)(ws + XB_B + WQKV_B);
  __bf16* qkvb   = (__bf16*)(ws + XB_B + WQKV_B + WPROJ_B);
  __bf16* attnb  = (__bf16*)(ws + XB_B + WQKV_B + WPROJ_B + QKV_B);
  __bf16* vtb    = (__bf16*)(ws + XB_B + WQKV_B + WPROJ_B + QKV_B + ATT_B);

  const int CVB = 2048, CVT = 256;
  cvt_f32_bf16<<<CVB, CVT, 0, stream>>>(x, xb, (long long)MTOK_ * DIM_);
  cvt_f32_bf16<<<CVB, CVT, 0, stream>>>(wq, wqkvb, (long long)DIM_ * DIM_);
  cvt_f32_bf16<<<CVB, CVT, 0, stream>>>(wk, wqkvb + (size_t)DIM_ * DIM_,
                                        (long long)NKV_ * DH_ * DIM_);
  cvt_f32_bf16<<<CVB, CVT, 0, stream>>>(wv, wqkvb + (size_t)VOFF_ * DIM_,
                                        (long long)NKV_ * DH_ * DIM_);
  cvt_f32_bf16<<<CVB, CVT, 0, stream>>>(wproj, wprojb, (long long)DIM_ * DIM_);

  // qkv = x @ [wq;wk;wv]^T      (M=4096, N=6144, K=4096)
  gemm_bf16_wmma<false><<<dim3(QKVN_ / 256, MTOK_ / 256), 256, 0, stream>>>(
      xb, wqkvb, qkvb, MTOK_, QKVN_, DIM_);

  // RoPE on Q and K slices
  {
    const long long total = (long long)MTOK_ * (NH_ + NKV_) * (DH_ / 2);
    rope_kernel<<<(int)((total + 255) / 256), 256, 0, stream>>>(qkvb, fcos, fsin);
  }

  // V transpose for B-fragment-friendly layout
  {
    const long long total = (long long)BB_ * NKV_ * DH_ * TT_;
    transpose_v<<<(int)((total + 255) / 256), 256, 0, stream>>>(qkvb, vtb);
  }

  // flash attention: grid (T/64, heads, batch), one wave per block
  attn_kernel<<<dim3(TT_ / 64, NH_, BB_), 32, 0, stream>>>(qkvb, vtb, attnb);

  // out = attn @ wproj^T        (M=4096, N=4096, K=4096), f32 epilogue
  gemm_bf16_wmma<true><<<dim3(DIM_ / 256, MTOK_ / 256), 256, 0, stream>>>(
      attnb, wprojb, d_out, MTOK_, DIM_, DIM_);
}
